// GQA_1365799600950
// MI455X (gfx1250) — matrix operand order repaired and checked by reading, not compiled, not run on hardware
//
#include <hip/hip_runtime.h>

typedef unsigned short u16;
typedef __attribute__((ext_vector_type(16))) __bf16 v16bf;
typedef __attribute__((ext_vector_type(8)))  __bf16 v8bf;
typedef __attribute__((ext_vector_type(8)))  float  v8f;

union Frag16 { v16bf v; v8bf h[2]; };
union V8U    { v8bf v; u16 u[8]; };

#define DMODEL 1024
#define NH     16
#define NKV    8
#define DH     64
#define TSEQ   2048
#define BATCH  2
#define MTOK   (BATCH*TSEQ)     /* 4096 tokens */
#define SCALE  0.125f           /* 64^-0.5 */

static __device__ __forceinline__ u16 f2bf(float f) {
  unsigned u = __float_as_uint(f);
  unsigned r = u + 0x7FFFu + ((u >> 16) & 1u);   // round-to-nearest-even
  return (u16)(r >> 16);
}

static __device__ __forceinline__ v8f wmma_bf16(v16bf a, v16bf b, v8f c) {
  return __builtin_amdgcn_wmma_f32_16x16x32_bf16(false, a, false, b, (short)0, c,
                                                 false, false);
}

// Async DMA: global -> LDS, 16 bytes per lane, tracked by ASYNCcnt.
static __device__ __forceinline__ void async_g2l_b128(unsigned lds_off,
                                                      unsigned vaddr_off,
                                                      unsigned long long sbase) {
  asm volatile("global_load_async_to_lds_b128 %0, %1, %2"
               :: "v"(lds_off), "v"(vaddr_off), "s"(sbase)
               : "memory");
}
static __device__ __forceinline__ void wait_asynccnt0() {
  asm volatile("s_wait_asynccnt 0x0" ::: "memory");
}

__global__ void cvt_f32_bf16(const float* __restrict__ in, u16* __restrict__ out, int n) {
  int i = blockIdx.x * blockDim.x + threadIdx.x;
  if (i < n) out[i] = f2bf(in[i]);
}

// C[M,N] = A[M,K](bf16) * W[N,K]^T(bf16). 256 thr = 8 waves.
// Block tile 256(M) x 64(N); wave tile 32 x 64 (8 accum tiles, 8 WMMA / k-step).
// W tile (64x32) staged in double-buffered LDS via async global->LDS DMA.
template <bool OUT_BF16>
__global__ __launch_bounds__(256) void gemm_bf16(const u16* __restrict__ A,
                                                 const u16* __restrict__ W,
                                                 void* __restrict__ Cout,
                                                 int M, int N, int K) {
  __shared__ __align__(32) u16 Wt[2][64 * 40];   // pitch 40 halves = 80B: conflict-free

  const int tid  = threadIdx.x;
  const int lane = tid & 31;
  const int wv   = tid >> 5;
  const int lh   = lane >> 4;
  const int l15  = lane & 15;
  const int m0   = blockIdx.y * 256 + wv * 32;
  const int n0   = blockIdx.x * 64;

  // one async b128 per thread per k-step covers the 64x32 W tile
  const int srow = tid >> 2;                     // 0..63  (n within tile)
  const int soff = (tid & 3) * 8;                // 0,8,16,24 (k within tile)
  const unsigned long long wbase = (unsigned long long)(W + (size_t)n0 * K);
  const unsigned lb[2] = { (unsigned)(size_t)&Wt[0][srow * 40 + soff],
                           (unsigned)(size_t)&Wt[1][srow * 40 + soff] };

  v8f acc[8] = {};
  const u16* arow0 = A + (size_t)(m0 + l15) * K + 8 * lh;
  const u16* arow1 = A + (size_t)(m0 + 16 + l15) * K + 8 * lh;

  async_g2l_b128(lb[0], (unsigned)((srow * K + soff) * 2), wbase);
  wait_asynccnt0();
  __syncthreads();

  int buf = 0;
  for (int k0 = 0; k0 < K; k0 += 32, buf ^= 1) {
    if (k0 + 32 < K)                             // overlap DMA of next tile
      async_g2l_b128(lb[buf ^ 1], (unsigned)((srow * K + soff + k0 + 32) * 2), wbase);

    Frag16 a0, a1;                               // A: K = k0+8lh+{0..7}, +16+8lh+{0..7}
    a0.h[0] = *(const v8bf*)(arow0 + k0);
    a0.h[1] = *(const v8bf*)(arow0 + k0 + 16);
    a1.h[0] = *(const v8bf*)(arow1 + k0);
    a1.h[1] = *(const v8bf*)(arow1 + k0 + 16);
#pragma unroll
    for (int nt = 0; nt < 4; ++nt) {             // B: lane=n, K = 8lh+{0..7}, +16+8lh+{0..7}
      const u16* wp = &Wt[buf][(nt * 16 + l15) * 40 + 8 * lh];
      Frag16 bf;
      bf.h[0] = *(const v8bf*)(wp);
      bf.h[1] = *(const v8bf*)(wp + 16);
      acc[nt]     = wmma_bf16(a0.v, bf.v, acc[nt]);
      acc[4 + nt] = wmma_bf16(a1.v, bf.v, acc[4 + nt]);
    }
    wait_asynccnt0();
    __syncthreads();
  }

#pragma unroll
  for (int half = 0; half < 2; ++half) {
#pragma unroll
    for (int nt = 0; nt < 4; ++nt) {
#pragma unroll
      for (int r = 0; r < 8; ++r) {              // C row = r + 8*laneHalf
        int row = m0 + half * 16 + 8 * lh + r;
        int col = n0 + nt * 16 + l15;
        float v = acc[half * 4 + nt][r];
        if (OUT_BF16) ((u16*)Cout)[(size_t)row * N + col] = f2bf(v);
        else          ((float*)Cout)[(size_t)row * N + col] = v;
      }
    }
  }
}

// Fused flash-attention: grid (32, 16 heads, 2 batch), 128 thr = 4 waves.
// Each wave: 16 query rows; block shares 64x64 K tile (async DMA to LDS) and
// a transposed V tile; P round-trips per-wave LDS for C->A fragment relayout.
__global__ __launch_bounds__(128) void attn_gqa(const u16* __restrict__ Qb,
                                                const u16* __restrict__ Kb,
                                                const u16* __restrict__ Vb,
                                                u16* __restrict__ Yb) {
  __shared__ __align__(32) u16 Ks[64 * 72];      // [s][d], pitch 144B: conflict-free
  __shared__ __align__(32) u16 Vt[64 * 72];      // [d][s] transposed
  __shared__ __align__(32) u16 Pb[4][16 * 72];   // per-wave P staging

  const int tid  = threadIdx.x;
  const int lane = tid & 31;
  const int wv   = tid >> 5;
  const int lh   = lane >> 4;
  const int l15  = lane & 15;

  const int t0 = blockIdx.x * 64;
  const int h  = blockIdx.y;
  const int b  = blockIdx.z;
  const int kv = h >> 1;                         // GROUP = 2
  const size_t kvbase = (size_t)kv * DH;

  // per-thread K-tile staging coords (4 async chunks)
  unsigned kvoff[4], kdst[4];
#pragma unroll
  for (int i = 0; i < 4; ++i) {
    int c = tid + i * 128, row = c >> 3, off = (c & 7) * 8;
    kvoff[i] = (unsigned)((row * (NKV * DH) + off) * 2);
    kdst[i]  = (unsigned)(size_t)&Ks[row * 72 + off];
  }

  Frag16 qf[2];
  {
    const u16* qp = Qb + (size_t)(b * TSEQ + t0 + wv * 16 + l15) * DMODEL + h * DH;
#pragma unroll
    for (int ks = 0; ks < 2; ++ks) {
      const u16* p = qp + ks * 32 + 8 * lh;
      qf[ks].h[0] = *(const v8bf*)(p);
      qf[ks].h[1] = *(const v8bf*)(p + 16);
    }
  }

  float mrow[8], lrow[8];
#pragma unroll
  for (int r = 0; r < 8; ++r) { mrow[r] = -3.0e38f; lrow[r] = 0.f; }
  v8f oacc[4] = {};

  for (int s0 = 0; s0 < TSEQ; s0 += 64) {
    // K tile: async DMA straight into LDS (overlaps the V transpose below)
    const unsigned long long ksad =
        (unsigned long long)(Kb + (size_t)(b * TSEQ + s0) * (NKV * DH) + kvbase);
#pragma unroll
    for (int i = 0; i < 4; ++i) async_g2l_b128(kdst[i], kvoff[i], ksad);

    // V tile: load rows, store transposed (needs VGPR round-trip)
    for (int c = tid; c < 512; c += 128) {
      int row = c >> 3, off = (c & 7) * 8;
      V8U vv;
      vv.v = *(const v8bf*)(Vb + (size_t)(b * TSEQ + s0 + row) * (NKV * DH) + kvbase + off);
#pragma unroll
      for (int j = 0; j < 8; ++j) Vt[(off + j) * 72 + row] = vv.u[j];
    }
    wait_asynccnt0();
    __syncthreads();

    // ---- S = Q * K^T ----
    v8f sf[4] = {};
#pragma unroll
    for (int nt = 0; nt < 4; ++nt) {
#pragma unroll
      for (int ks = 0; ks < 2; ++ks) {
        const u16* kp = &Ks[(nt * 16 + l15) * 72 + ks * 32 + 8 * lh];
        Frag16 bk;
        bk.h[0] = *(const v8bf*)(kp);
        bk.h[1] = *(const v8bf*)(kp + 16);
        sf[nt] = wmma_bf16(qf[ks].v, bk.v, sf[nt]);
      }
    }
#pragma unroll
    for (int nt = 0; nt < 4; ++nt) sf[nt] = sf[nt] * SCALE;

    // ---- online softmax (row m = r + 8*laneHalf across 16 lanes) ----
    float alpha[8];
#pragma unroll
    for (int r = 0; r < 8; ++r) {
      float mx = sf[0][r];
#pragma unroll
      for (int nt = 1; nt < 4; ++nt) mx = fmaxf(mx, sf[nt][r]);
#pragma unroll
      for (int d = 1; d < 16; d <<= 1) mx = fmaxf(mx, __shfl_xor(mx, d, 32));
      float mn = fmaxf(mrow[r], mx);
      alpha[r] = __expf(mrow[r] - mn);
      mrow[r]  = mn;
    }
    float rs[8];
#pragma unroll
    for (int r = 0; r < 8; ++r) rs[r] = 0.f;
#pragma unroll
    for (int nt = 0; nt < 4; ++nt) {
#pragma unroll
      for (int r = 0; r < 8; ++r) {
        float p = __expf(sf[nt][r] - mrow[r]);
        rs[r] += p;
        Pb[wv][(r + 8 * lh) * 72 + nt * 16 + l15] = f2bf(p);
      }
    }
    v8f av;
#pragma unroll
    for (int r = 0; r < 8; ++r) {
#pragma unroll
      for (int d = 1; d < 16; d <<= 1) rs[r] += __shfl_xor(rs[r], d, 32);
      lrow[r] = lrow[r] * alpha[r] + rs[r];
      av[r]   = alpha[r];
    }
#pragma unroll
    for (int dt = 0; dt < 4; ++dt) oacc[dt] = oacc[dt] * av;

    // ---- O += P * V ----
#pragma unroll
    for (int dt = 0; dt < 4; ++dt) {
#pragma unroll
      for (int ks = 0; ks < 2; ++ks) {
        const u16* pp = &Pb[wv][l15 * 72 + ks * 32 + 8 * lh];
        Frag16 pa;
        pa.h[0] = *(const v8bf*)(pp);
        pa.h[1] = *(const v8bf*)(pp + 16);
        const u16* vp = &Vt[(dt * 16 + l15) * 72 + ks * 32 + 8 * lh];
        Frag16 vb;
        vb.h[0] = *(const v8bf*)(vp);
        vb.h[1] = *(const v8bf*)(vp + 16);
        oacc[dt] = wmma_bf16(pa.v, vb.v, oacc[dt]);
      }
    }
    __syncthreads();
  }

  v8f inv;
#pragma unroll
  for (int r = 0; r < 8; ++r) inv[r] = 1.0f / lrow[r];
#pragma unroll
  for (int dt = 0; dt < 4; ++dt) {
    v8f ov = oacc[dt] * inv;
#pragma unroll
    for (int r = 0; r < 8; ++r) {
      int row = b * TSEQ + t0 + wv * 16 + 8 * lh + r;
      Yb[(size_t)row * DMODEL + h * DH + dt * 16 + l15] = f2bf(ov[r]);
    }
  }
}

extern "C" void kernel_launch(void* const* d_in, const int* in_sizes, int n_in,
                              void* d_out, int out_size, void* d_ws, size_t ws_size,
                              hipStream_t stream) {
  (void)in_sizes; (void)n_in; (void)out_size; (void)ws_size;
  const float* x  = (const float*)d_in[0];
  const float* Wq = (const float*)d_in[1];
  const float* Wk = (const float*)d_in[2];
  const float* Wv = (const float*)d_in[3];
  const float* Wp = (const float*)d_in[4];

  u16* ws  = (u16*)d_ws;
  u16* xb  = ws;
  u16* wqb = xb  + (size_t)MTOK * DMODEL;
  u16* wkb = wqb + (size_t)DMODEL * DMODEL;
  u16* wvb = wkb + (size_t)(NKV * DH) * DMODEL;
  u16* wpb = wvb + (size_t)(NKV * DH) * DMODEL;
  u16* qb  = wpb + (size_t)DMODEL * DMODEL;
  u16* kb  = qb  + (size_t)MTOK * DMODEL;
  u16* vb  = kb  + (size_t)MTOK * (NKV * DH);
  u16* yb  = vb  + (size_t)MTOK * (NKV * DH);

  cvt_f32_bf16<<<(MTOK * DMODEL + 255) / 256, 256, 0, stream>>>(x, xb, MTOK * DMODEL);
  cvt_f32_bf16<<<(DMODEL * DMODEL + 255) / 256, 256, 0, stream>>>(Wq, wqb, DMODEL * DMODEL);
  cvt_f32_bf16<<<(NKV * DH * DMODEL + 255) / 256, 256, 0, stream>>>(Wk, wkb, NKV * DH * DMODEL);
  cvt_f32_bf16<<<(NKV * DH * DMODEL + 255) / 256, 256, 0, stream>>>(Wv, wvb, NKV * DH * DMODEL);
  cvt_f32_bf16<<<(DMODEL * DMODEL + 255) / 256, 256, 0, stream>>>(Wp, wpb, DMODEL * DMODEL);

  gemm_bf16<true><<<dim3(DMODEL / 64, MTOK / 256), 256, 0, stream>>>(xb, wqb, qb, MTOK, DMODEL, DMODEL);
  gemm_bf16<true><<<dim3((NKV * DH) / 64, MTOK / 256), 256, 0, stream>>>(xb, wkb, kb, MTOK, NKV * DH, DMODEL);
  gemm_bf16<true><<<dim3((NKV * DH) / 64, MTOK / 256), 256, 0, stream>>>(xb, wvb, vb, MTOK, NKV * DH, DMODEL);

  attn_gqa<<<dim3(TSEQ / 64, NH, BATCH), 128, 0, stream>>>(qb, kb, vb, yb);

  gemm_bf16<false><<<dim3(DMODEL / 64, MTOK / 256), 256, 0, stream>>>(yb, wpb, d_out, MTOK, DMODEL, DMODEL);
}
